// TernaryLinear_49065706389531
// MI455X (gfx1250) — matrix operand order repaired and checked by reading, not compiled, not run on hardware
//
#include <hip/hip_runtime.h>

// ---------------------------------------------------------------------------
// TernaryLinear: Y = X @ ternary(W)^T + bias   (M=8192, N=4096, K=4096, fp32)
//
// Compute-bound (AI ~818 FLOP/B): use the bf16 WMMA pipe. Ternary weights are
// exact in bf16; activations use a hi/lo bf16 split with two
// v_wmma_f32_16x16x32_bf16 per tile into one f32 accumulator (near-fp32
// accuracy at bf16 speed).
//
// Fast path (ws >= 160MB): convert once (memory-bound, ~22us) into d_ws, then
// run a conversion-free WMMA GEMM whose staging uses async global->LDS loads.
// Fallback path: fused conversion GEMM (no workspace needed).
// ---------------------------------------------------------------------------

typedef __attribute__((ext_vector_type(16))) __bf16 v16bf;
typedef __attribute__((ext_vector_type(8)))  float  v8f;
typedef int v4i __attribute__((vector_size(16)));   // matches async-LDS builtin

#define TILE_M 128
#define TILE_N 128
#define TILE_K 32
#define LDS_K  40            // padded K stride (elements) to spread LDS banks
#define KDIM   4096
#define NDIM   4096
#define NKT    (KDIM / TILE_K)

struct alignas(16) F4  { float x, y, z, w; };
struct alignas(8)  U2  { unsigned int x, y; };
struct alignas(16) U4  { unsigned int x, y, z, w; };
struct U4x2 { U4 a, b; };    // 32 bytes == v16bf

// ---------------- scalar conversion helpers --------------------------------

static __device__ __forceinline__ unsigned int bf16rne(float f) {
    unsigned int u = __builtin_bit_cast(unsigned int, f);
    return (u + 0x7FFFu + ((u >> 16) & 1u)) >> 16;   // round-to-nearest-even
}
static __device__ __forceinline__ float from_lo(unsigned int p) {
    return __builtin_bit_cast(float, p << 16);
}
static __device__ __forceinline__ float from_hi(unsigned int p) {
    return __builtin_bit_cast(float, p & 0xFFFF0000u);
}
// pack two floats to packed bf16 (a -> low half, b -> high half), RNE
static __device__ __forceinline__ unsigned int pk_bf16(float a, float b) {
#if __has_builtin(__builtin_amdgcn_cvt_pk_bf16_f32)
    return __builtin_bit_cast(unsigned int, __builtin_amdgcn_cvt_pk_bf16_f32(a, b));
#else
    return bf16rne(a) | (bf16rne(b) << 16);
#endif
}
// ternary quantize -> bf16 bits of {-1, 0, +1}
static __device__ __forceinline__ unsigned int ternq(float w) {
    unsigned int u = __builtin_bit_cast(unsigned int, w);
    return (__builtin_fabsf(w) > 0.05f) ? (((u >> 16) & 0x8000u) | 0x3F80u) : 0u;
}
static __device__ __forceinline__ unsigned int pk_tern(float a, float b) {
    unsigned int qa = ternq(a), qb = ternq(b);
#if __has_builtin(__builtin_amdgcn_perm)
    return __builtin_amdgcn_perm(qb, qa, 0x05040100u);   // qa -> low16, qb -> high16
#else
    return qa | (qb << 16);
#endif
}
static __device__ __forceinline__ v16bf load_frag(const unsigned short* p0,
                                                  const unsigned short* p1) {
    U4x2 t;
    t.a = *(const U4*)p0;
    t.b = *(const U4*)p1;
    return __builtin_bit_cast(v16bf, t);
}

// ---------------- async global -> LDS staging (CDNA5) ----------------------

#if __has_builtin(__builtin_amdgcn_global_load_async_to_lds_b128)
#define TL_ASYNC_STAGE 1
static __device__ __forceinline__ void async_b128(const unsigned short* g,
                                                  unsigned short* l) {
    __builtin_amdgcn_global_load_async_to_lds_b128(
        (__attribute__((address_space(1))) v4i*)(g),
        (__attribute__((address_space(3))) v4i*)(l), 0, 0);
}
static __device__ __forceinline__ void wait_async0() {
#if __has_builtin(__builtin_amdgcn_s_wait_asynccnt)
    __builtin_amdgcn_s_wait_asynccnt(0);
#else
    asm volatile("s_wait_asynccnt 0" ::: "memory");
#endif
}
#else
#define TL_ASYNC_STAGE 0
#endif

// ---------------- prep kernels (memory-bound, run once per call) -----------

__global__ __launch_bounds__(256)
void convert_x_kernel(const float* __restrict__ X,
                      unsigned int* __restrict__ Xhi,
                      unsigned int* __restrict__ Xlo, long n4) {
    const long i = (long)blockIdx.x * 256 + threadIdx.x;   // one per 4 floats
    if (i >= n4) return;
    const F4 v = *(const F4*)(X + i * 4);
    const unsigned int h0 = pk_bf16(v.x, v.y);
    const unsigned int h1 = pk_bf16(v.z, v.w);
    U2 hp; hp.x = h0; hp.y = h1;
    U2 lp;
    lp.x = pk_bf16(v.x - from_lo(h0), v.y - from_hi(h0));
    lp.y = pk_bf16(v.z - from_lo(h1), v.w - from_hi(h1));
    *(U2*)(Xhi + i * 2) = hp;
    *(U2*)(Xlo + i * 2) = lp;
}

__global__ __launch_bounds__(256)
void convert_w_kernel(const float* __restrict__ W,
                      unsigned int* __restrict__ Wq, long n4) {
    const long i = (long)blockIdx.x * 256 + threadIdx.x;   // one per 4 floats
    if (i >= n4) return;
    const F4 v = *(const F4*)(W + i * 4);
    U2 q; q.x = pk_tern(v.x, v.y); q.y = pk_tern(v.z, v.w);
    *(U2*)(Wq + i * 2) = q;
}

// ---------------- fast path: conversion-free WMMA GEMM ---------------------

__global__ __launch_bounds__(256)
void ternary_gemm_bf16(const unsigned short* __restrict__ Xhi,
                       const unsigned short* __restrict__ Xlo,
                       const unsigned short* __restrict__ Wq,
                       const float* __restrict__ bias,
                       float* __restrict__ out)
{
    __shared__ unsigned short sAhi[2][TILE_M * LDS_K];
    __shared__ unsigned short sAlo[2][TILE_M * LDS_K];
    __shared__ unsigned short sB  [2][TILE_N * LDS_K];

    const int tid   = threadIdx.x;
    const int lane  = tid & 31;
    const int wave  = tid >> 5;
    const int waveM = wave >> 1;     // 0..3
    const int waveN = wave & 1;      // 0..1
    const int bm    = blockIdx.y * TILE_M;
    const int bn    = blockIdx.x * TILE_N;
    const int lm    = lane & 15;
    const int lh    = lane >> 4;

    // staging: 16B per thread per row-pass; 64 rows/pass, 2 passes
    const int srow = tid >> 2;          // 0..63
    const int scol = (tid & 3) << 3;    // 0,8,16,24 (bf16 elements)

    v8f acc[2][4];
#pragma unroll
    for (int mf = 0; mf < 2; ++mf)
#pragma unroll
        for (int nf = 0; nf < 4; ++nf)
            acc[mf][nf] = (v8f){0.f, 0.f, 0.f, 0.f, 0.f, 0.f, 0.f, 0.f};

    auto compute = [&](int buf) {
        v16bf afh[2], afl[2], bfr[4];
#pragma unroll
        for (int mf = 0; mf < 2; ++mf) {
            // A 16x32 bf16: lanes 0-15 -> K[lh*8..+7] and K[16+lh*8..+7]
            const int row = waveM * 32 + mf * 16 + lm;
            const unsigned short* p = &sAhi[buf][row * LDS_K + lh * 8];
            afh[mf] = load_frag(p, p + 16);
            const unsigned short* q = &sAlo[buf][row * LDS_K + lh * 8];
            afl[mf] = load_frag(q, q + 16);
        }
#pragma unroll
        for (int nf = 0; nf < 4; ++nf) {
            // B 32x16 bf16: lane = column lm, K[lh*8..+7] and K[16+lh*8..+7]
            const int row = waveN * 64 + nf * 16 + lm;
            const unsigned short* p = &sB[buf][row * LDS_K + lh * 8];
            bfr[nf] = load_frag(p, p + 16);
        }
#pragma unroll
        for (int mf = 0; mf < 2; ++mf)
#pragma unroll
            for (int nf = 0; nf < 4; ++nf) {
                acc[mf][nf] = __builtin_amdgcn_wmma_f32_16x16x32_bf16(
                    false, afh[mf], false, bfr[nf], (short)0, acc[mf][nf], false, false);
                acc[mf][nf] = __builtin_amdgcn_wmma_f32_16x16x32_bf16(
                    false, afl[mf], false, bfr[nf], (short)0, acc[mf][nf], false, false);
            }
    };

#if TL_ASYNC_STAGE
    auto issue = [&](int kt, int buf) {
        const int kk = kt * TILE_K + scol;
#pragma unroll
        for (int p = 0; p < 2; ++p) {
            const int r = srow + (p << 6);
            const int loff = r * LDS_K + scol;
            async_b128(Xhi + (size_t)(bm + r) * KDIM + kk, &sAhi[buf][loff]);
            async_b128(Xlo + (size_t)(bm + r) * KDIM + kk, &sAlo[buf][loff]);
            async_b128(Wq  + (size_t)(bn + r) * KDIM + kk, &sB  [buf][loff]);
        }
    };

    issue(0, 0);
    wait_async0();
    __syncthreads();
    for (int kt = 0; kt < NKT; ++kt) {
        const int buf = kt & 1;
        if (kt + 1 < NKT) issue(kt + 1, buf ^ 1);  // async writes overlap compute
        compute(buf);
        wait_async0();                              // our LDS writes landed
        __syncthreads();                            // publish buf^1 to all waves
    }
#else
    U4 rh[2], rl[2], rw[2];
    auto gload = [&](int kt) {
        const int kk = kt * TILE_K + scol;
#pragma unroll
        for (int p = 0; p < 2; ++p) {
            const int r = srow + (p << 6);
            rh[p] = *(const U4*)(Xhi + (size_t)(bm + r) * KDIM + kk);
            rl[p] = *(const U4*)(Xlo + (size_t)(bm + r) * KDIM + kk);
            rw[p] = *(const U4*)(Wq  + (size_t)(bn + r) * KDIM + kk);
        }
    };
    auto store = [&](int buf) {
#pragma unroll
        for (int p = 0; p < 2; ++p) {
            const int loff = (srow + (p << 6)) * LDS_K + scol;
            *(U4*)&sAhi[buf][loff] = rh[p];
            *(U4*)&sAlo[buf][loff] = rl[p];
            *(U4*)&sB  [buf][loff] = rw[p];
        }
    };
    gload(0);
    store(0);
    __syncthreads();
    for (int kt = 0; kt < NKT; ++kt) {
        const int buf = kt & 1;
        if (kt + 1 < NKT) gload(kt + 1);
        compute(buf);
        __syncthreads();
        if (kt + 1 < NKT) store(buf ^ 1);
        __syncthreads();
    }
#endif

    // Epilogue: C layout -> lane (lm = N, lh = M half), VGPR j -> M = j + 8*lh
#pragma unroll
    for (int nf = 0; nf < 4; ++nf) {
        const int gn = bn + waveN * 64 + nf * 16 + lm;
        const float bv = bias[gn];
#pragma unroll
        for (int mf = 0; mf < 2; ++mf) {
            const int gm = bm + waveM * 32 + mf * 16 + lh * 8;
#pragma unroll
            for (int j = 0; j < 8; ++j)
                out[(size_t)(gm + j) * NDIM + gn] = acc[mf][nf][j] + bv;
        }
    }
}

// ---------------- fallback path: fused conversion GEMM ---------------------

__global__ __launch_bounds__(256)
void ternary_linear_fused(const float* __restrict__ X,
                          const float* __restrict__ W,
                          const float* __restrict__ bias,
                          float* __restrict__ out)
{
    __shared__ unsigned short sAhi[2][TILE_M * LDS_K];
    __shared__ unsigned short sAlo[2][TILE_M * LDS_K];
    __shared__ unsigned short sB  [2][TILE_N * LDS_K];

    const int tid   = threadIdx.x;
    const int lane  = tid & 31;
    const int wave  = tid >> 5;
    const int waveM = wave >> 1;
    const int waveN = wave & 1;
    const int bm    = blockIdx.y * TILE_M;
    const int bn    = blockIdx.x * TILE_N;
    const int lm    = lane & 15;
    const int lh    = lane >> 4;

    const int tr = tid >> 3;         // 0..31
    const int tc = (tid & 7) << 2;   // 0,4,...,28

    v8f acc[2][4];
#pragma unroll
    for (int mf = 0; mf < 2; ++mf)
#pragma unroll
        for (int nf = 0; nf < 4; ++nf)
            acc[mf][nf] = (v8f){0.f, 0.f, 0.f, 0.f, 0.f, 0.f, 0.f, 0.f};

    F4 rx[4], rw[4];

    auto gload = [&](int kt) {
        const int kk = kt * TILE_K + tc;
#pragma unroll
        for (int p = 0; p < 4; ++p) {
            const int r = tr + p * 32;
            rx[p] = *(const F4*)(X + (size_t)(bm + r) * KDIM + kk);
            rw[p] = *(const F4*)(W + (size_t)(bn + r) * KDIM + kk);
        }
    };
    auto stage = [&](int buf) {
#pragma unroll
        for (int p = 0; p < 4; ++p) {
            const int off = (tr + p * 32) * LDS_K + tc;
            const unsigned int h0 = pk_bf16(rx[p].x, rx[p].y);
            const unsigned int h1 = pk_bf16(rx[p].z, rx[p].w);
            U2 hp; hp.x = h0; hp.y = h1;
            U2 lp;
            lp.x = pk_bf16(rx[p].x - from_lo(h0), rx[p].y - from_hi(h0));
            lp.y = pk_bf16(rx[p].z - from_lo(h1), rx[p].w - from_hi(h1));
            *(U2*)&sAhi[buf][off] = hp;
            *(U2*)&sAlo[buf][off] = lp;
            U2 qp; qp.x = pk_tern(rw[p].x, rw[p].y); qp.y = pk_tern(rw[p].z, rw[p].w);
            *(U2*)&sB[buf][off] = qp;
        }
    };

    gload(0);
    stage(0);
    __syncthreads();

    for (int kt = 0; kt < NKT; ++kt) {
        const int buf = kt & 1;
        if (kt + 1 < NKT) gload(kt + 1);

        v16bf afh[2], afl[2], bfr[4];
#pragma unroll
        for (int mf = 0; mf < 2; ++mf) {
            const int row = waveM * 32 + mf * 16 + lm;
            const unsigned short* p = &sAhi[buf][row * LDS_K + lh * 8];
            afh[mf] = load_frag(p, p + 16);
            const unsigned short* q = &sAlo[buf][row * LDS_K + lh * 8];
            afl[mf] = load_frag(q, q + 16);
        }
#pragma unroll
        for (int nf = 0; nf < 4; ++nf) {
            const int row = waveN * 64 + nf * 16 + lm;
            const unsigned short* p = &sB[buf][row * LDS_K + lh * 8];
            bfr[nf] = load_frag(p, p + 16);
        }
#pragma unroll
        for (int mf = 0; mf < 2; ++mf)
#pragma unroll
            for (int nf = 0; nf < 4; ++nf) {
                acc[mf][nf] = __builtin_amdgcn_wmma_f32_16x16x32_bf16(
                    false, afh[mf], false, bfr[nf], (short)0, acc[mf][nf], false, false);
                acc[mf][nf] = __builtin_amdgcn_wmma_f32_16x16x32_bf16(
                    false, afl[mf], false, bfr[nf], (short)0, acc[mf][nf], false, false);
            }

        __syncthreads();
        if (kt + 1 < NKT) stage(buf ^ 1);
        __syncthreads();
    }

#pragma unroll
    for (int nf = 0; nf < 4; ++nf) {
        const int gn = bn + waveN * 64 + nf * 16 + lm;
        const float bv = bias[gn];
#pragma unroll
        for (int mf = 0; mf < 2; ++mf) {
            const int gm = bm + waveM * 32 + mf * 16 + lh * 8;
#pragma unroll
            for (int j = 0; j < 8; ++j)
                out[(size_t)(gm + j) * NDIM + gn] = acc[mf][nf][j] + bv;
        }
    }
}

// ---------------- launch ---------------------------------------------------

extern "C" void kernel_launch(void* const* d_in, const int* in_sizes, int n_in,
                              void* d_out, int out_size, void* d_ws, size_t ws_size,
                              hipStream_t stream) {
    const float* X = (const float*)d_in[0];   // [tokens, 4096]
    const float* W = (const float*)d_in[1];   // [4096, 4096]
    const float* B = (const float*)d_in[2];   // [4096]
    float* out = (float*)d_out;               // [tokens, 4096]

    const long tokens = in_sizes[0] / KDIM;   // 8192
    const size_t xElems = (size_t)tokens * KDIM;
    const size_t wElems = (size_t)NDIM * KDIM;
    const size_t need   = xElems * 2 * 2 + wElems * 2;  // Xhi + Xlo + Wq (bf16)

    dim3 grid(NDIM / TILE_N, (unsigned)(tokens / TILE_M));

    if (d_ws && ws_size >= need) {
        unsigned short* xhi = (unsigned short*)d_ws;
        unsigned short* xlo = xhi + xElems;
        unsigned short* wq  = xlo + xElems;
        const long n4x = (long)(xElems / 4);
        const long n4w = (long)(wElems / 4);
        convert_x_kernel<<<(unsigned)((n4x + 255) / 256), 256, 0, stream>>>(
            X, (unsigned int*)xhi, (unsigned int*)xlo, n4x);
        convert_w_kernel<<<(unsigned)((n4w + 255) / 256), 256, 0, stream>>>(
            W, (unsigned int*)wq, n4w);
        ternary_gemm_bf16<<<grid, 256, 0, stream>>>(xhi, xlo, wq, B, out);
    } else {
        ternary_linear_fused<<<grid, 256, 0, stream>>>(X, W, B, out);
    }
}
